// GNNModel_17317308137513
// MI455X (gfx1250) — matrix operand order repaired and checked by reading, not compiled, not run on hardware
//
#include <hip/hip_runtime.h>

typedef __attribute__((ext_vector_type(16))) _Float16     v16h;
typedef __attribute__((ext_vector_type(8)))  float        v8f;
typedef __attribute__((ext_vector_type(4)))  unsigned int u32x4;

#define EPS_BN 1e-5f

// ---------------------------------------------------------------------------
// Gather / concat kernels (write f16 concat buffers, zero the K-padding)
// ---------------------------------------------------------------------------
__global__ void gather_edge_in(const float* __restrict__ x, const float* __restrict__ ea,
                               const int* __restrict__ row, const int* __restrict__ col,
                               long E, _Float16* __restrict__ e0 /*[E,32]*/,
                               _Float16* __restrict__ n1in /*[E,96]*/)
{
    long e = (long)blockIdx.x * blockDim.x + threadIdx.x;
    if (e >= E) return;
    long r = row[e], c = col[e];
    _Float16* pe = e0 + e * 32;
    _Float16* pn = n1in + e * 96;
#pragma unroll
    for (int j = 0; j < 9; ++j) {
        float v = x[r * 9 + j];
        pe[j] = (_Float16)v;
        pn[j] = (_Float16)v;               // node1 concat cols 0..8 = x[row]
    }
#pragma unroll
    for (int j = 0; j < 9; ++j) pe[9 + j] = (_Float16)x[c * 9 + j];
#pragma unroll
    for (int j = 0; j < 12; ++j) pe[18 + j] = (_Float16)ea[e * 12 + j];
    pe[30] = (_Float16)0.f; pe[31] = (_Float16)0.f;
#pragma unroll
    for (int j = 73; j < 96; ++j) pn[j] = (_Float16)0.f;   // pad 73..95
}

__global__ void gather_node_in(const float* __restrict__ x, long N,
                               _Float16* __restrict__ n2in /*[N,96]*/)
{
    long n = (long)blockIdx.x * blockDim.x + threadIdx.x;
    if (n >= N) return;
    _Float16* p = n2in + n * 96;
#pragma unroll
    for (int j = 0; j < 9; ++j) p[j] = (_Float16)x[n * 9 + j];
#pragma unroll
    for (int j = 73; j < 96; ++j) p[j] = (_Float16)0.f;
}

// ---------------------------------------------------------------------------
// Per-column batch statistics (sum, sum of squares) for training-mode BN
// grid = (64, nfeat); block = 256
// ---------------------------------------------------------------------------
__global__ void stats_cols(const _Float16* __restrict__ A, long rows, int stride,
                           float* __restrict__ sums, float* __restrict__ sumsq)
{
    int f = blockIdx.y;
    int tid = threadIdx.x;
    long r = (long)blockIdx.x * blockDim.x + tid;
    long step = (long)gridDim.x * blockDim.x;
    float s = 0.f, q = 0.f;
    for (; r < rows; r += step) {
        float v = (float)A[r * (long)stride + f];
        s += v; q += v * v;
    }
    __shared__ float ls[256];
    __shared__ float lq[256];
    ls[tid] = s; lq[tid] = q;
    __syncthreads();
    for (int off = 128; off > 0; off >>= 1) {
        if (tid < off) { ls[tid] += ls[tid + off]; lq[tid] += lq[tid + off]; }
        __syncthreads();
    }
    if (tid == 0) {
        atomicAdd(&sums[f], ls[0]);
        atomicAdd(&sumsq[f], lq[0]);
    }
}

// ---------------------------------------------------------------------------
// Fold BN into weights, emit f16 weights pre-packed in the WMMA B-fragment
// layout (v_wmma_f32_16x16x32_f16): dword index = ((kb*4+t)*32+lane)*8+v,
// lanes 0..15 hold K=kb*32+{0..7,16..23} (2 per VGPR), lanes 16..31 hold K=+{8..15,24..31}.
// Also compute folded bias' = c + shift @ W.  One block, 256 threads.
// ---------------------------------------------------------------------------
__global__ void fold_bn(const float* __restrict__ W /*[K,64]*/, const float* __restrict__ c,
                        const float* __restrict__ g, const float* __restrict__ be,
                        const float* __restrict__ sums, const float* __restrict__ sumsq,
                        float rowsInv, int K, int Kpad,
                        unsigned int* __restrict__ Wp, float* __restrict__ bias)
{
    __shared__ float scale[96];
    __shared__ float shift[96];
    int tid = threadIdx.x;
    if (tid < K) {
        float m = sums[tid] * rowsInv;
        float v = sumsq[tid] * rowsInv - m * m;
        float rs = rsqrtf(v + EPS_BN);
        float sc = g[tid] * rs;
        scale[tid] = sc;
        shift[tid] = be[tid] - m * sc;
    }
    __syncthreads();
    if (tid < 64) {
        float b = c[tid];
        for (int k = 0; k < K; ++k) b += shift[k] * W[(long)k * 64 + tid];
        bias[tid] = b;
    }
    int total = (Kpad >> 5) << 10;   // (Kpad/32)*4*32*8 dwords
    for (int i = tid; i < total; i += 256) {
        int v8   = i & 7;
        int lane = (i >> 3) & 31;
        int t    = (i >> 8) & 3;
        int kb   = i >> 10;
        int k0   = (kb << 5) + ((lane < 16) ? 0 : 8) + (v8 << 1) + ((v8 >= 4) ? 8 : 0);
        int colw = (t << 4) + (lane & 15);
        float w0 = (k0     < K) ? scale[k0]     * W[(long)k0 * 64 + colw]       : 0.f;
        float w1 = (k0 + 1 < K) ? scale[k0 + 1] * W[(long)(k0 + 1) * 64 + colw] : 0.f;
        _Float16 h0 = (_Float16)w0, h1 = (_Float16)w1;
        unsigned short u0 = __builtin_bit_cast(unsigned short, h0);
        unsigned short u1 = __builtin_bit_cast(unsigned short, h1);
        Wp[i] = (unsigned int)u0 | ((unsigned int)u1 << 16);
    }
}

// ---------------------------------------------------------------------------
// WMMA GEMM: [rows, Kpad] f16  x  [Kpad, 64] packed f16  (+bias, opt ReLU)
// 256 threads = 8 wave32 waves, one 16-row tile per wave, 4 n-tiles of 16.
// Either stores f16 to `out` (stride/colOff) or atomically scatter-adds f32
// rows into `accum[idx[row]*64 + col]` (fused segment-sum).
// ---------------------------------------------------------------------------
__global__ __launch_bounds__(256) void gemm64(
    const _Float16* __restrict__ A, long rows, int kpad,
    const unsigned int* __restrict__ Wp, const float* __restrict__ bias,
    _Float16* __restrict__ out, int outStride, int outColOff, int relu,
    float* __restrict__ accum, const int* __restrict__ idx)
{
    __shared__ __attribute__((aligned(16))) unsigned int sW[3 * 1024]; // up to Kpad=96
    __shared__ float sB[64];
    int tid = threadIdx.x;
    int nkb = kpad >> 5;
    int nW = nkb << 10;
    for (int i = tid; i < nW; i += 256) sW[i] = Wp[i];
    if (tid < 64) sB[tid] = bias[tid];
    __syncthreads();

    int wave = tid >> 5;
    int lane = tid & 31;
    long tile = ((long)blockIdx.x * 8 + wave) * 16;
    if (tile >= rows) return;

    long rowA = tile + (lane & 15);
    if (rowA >= rows) rowA = rows - 1;            // clamp; garbage rows never stored
    const _Float16* arow = A + rowA * (long)kpad;
    int koff = (lane & 16) ? 8 : 0;               // A fragment K-phase per half-wave

    v8f acc[4] = {};
    for (int kb = 0; kb < nkb; ++kb) {
        int k = kb << 5;
        v16h a;
        *((u32x4*)&a)       = *(const u32x4*)(arow + k + koff);        // K +0..7
        *(((u32x4*)&a) + 1) = *(const u32x4*)(arow + k + koff + 16);   // K +16..23
        if (kb + 1 < nkb) __builtin_prefetch(arow + k + 32 + koff, 0, 1);
        const unsigned int* wbase = sW + (kb << 10) + (lane << 3);
#pragma unroll
        for (int t = 0; t < 4; ++t) {
            v16h b;
            const unsigned int* wb = wbase + (t << 8);
            *((u32x4*)&b)       = *(const u32x4*)wb;
            *(((u32x4*)&b) + 1) = *(const u32x4*)(wb + 4);
            acc[t] = __builtin_amdgcn_wmma_f32_16x16x32_f16(
                false, a, false, b, (short)0, acc[t], false, false);
        }
    }

    int colbase = lane & 15;
    int mofs = (lane & 16) ? 8 : 0;
#pragma unroll
    for (int t = 0; t < 4; ++t) {
        int ocol = (t << 4) + colbase;
        float bcol = sB[ocol];
#pragma unroll
        for (int r = 0; r < 8; ++r) {
            long orow = tile + mofs + r;
            if (orow >= rows) continue;
            float v = acc[t][r] + bcol;
            if (relu) v = v > 0.f ? v : 0.f;      // LeakyReLU slope 0
            if (idx) {
                long seg = idx[orow];
                atomicAdd(&accum[seg * 64 + ocol], v);
            } else {
                out[orow * (long)outStride + outColOff + ocol] = (_Float16)v;
            }
        }
    }
}

// ---------------------------------------------------------------------------
// Segment helpers
// ---------------------------------------------------------------------------
__global__ void count_seg(const int* __restrict__ idx, long n, float* __restrict__ cnt)
{
    long i = (long)blockIdx.x * blockDim.x + threadIdx.x;
    if (i < n) atomicAdd(&cnt[idx[i]], 1.f);
}

__global__ void div_seg(const float* __restrict__ sum, const float* __restrict__ cnt,
                        long nseg, _Float16* __restrict__ dst, int dstStride, int dstColOff)
{
    long i = (long)blockIdx.x * blockDim.x + threadIdx.x;
    if (i >= nseg * 64) return;
    long s = i >> 6;
    int f = (int)(i & 63);
    float c = cnt[s];
    if (c < 1.f) c = 1.f;
    dst[s * (long)dstStride + dstColOff + f] = (_Float16)(sum[i] / c);
}

// ---------------------------------------------------------------------------
// Prediction head: logits = u @ predW + predb; softmax over 6
// ---------------------------------------------------------------------------
__global__ void pred_softmax(const _Float16* __restrict__ u, const float* __restrict__ Wp,
                             const float* __restrict__ bp, float* __restrict__ outp, int G)
{
    int g = blockIdx.x * blockDim.x + threadIdx.x;
    if (g >= G) return;
    float logit[6];
#pragma unroll
    for (int o = 0; o < 6; ++o) {
        float s = bp[o];
        for (int k = 0; k < 64; ++k) s += (float)u[(long)g * 64 + k] * Wp[k * 6 + o];
        logit[o] = s;
    }
    float mx = logit[0];
#pragma unroll
    for (int o = 1; o < 6; ++o) mx = fmaxf(mx, logit[o]);
    float den = 0.f, ex[6];
#pragma unroll
    for (int o = 0; o < 6; ++o) { ex[o] = __expf(logit[o] - mx); den += ex[o]; }
#pragma unroll
    for (int o = 0; o < 6; ++o) outp[(long)g * 6 + o] = ex[o] / den;
}

// ---------------------------------------------------------------------------
// Host orchestration
// ---------------------------------------------------------------------------
extern "C" void kernel_launch(void* const* d_in, const int* in_sizes, int n_in,
                              void* d_out, int out_size, void* d_ws, size_t ws_size,
                              hipStream_t stream)
{
    const float* x     = (const float*)d_in[0];
    const float* ea    = (const float*)d_in[1];
    const int*   eidx  = (const int*)d_in[2];
    const int*   batch = (const int*)d_in[3];
    const long N = in_sizes[0] / 9;
    const long E = in_sizes[1] / 12;
    const int  G = 256;
    const int* row = eidx;
    const int* col = eidx + E;

    // ---- workspace layout ----
    char* ws = (char*)d_ws;
    size_t off = 0;
    auto take = [&](size_t b) -> char* {
        char* p = ws + off;
        off = (off + b + 255) & ~(size_t)255;
        return p;
    };
    float*        stats   = (float*)take(12 * 192 * sizeof(float));         // 12 layers x (96 sum + 96 sq)
    unsigned int* wpAll   = (unsigned int*)take(12 * 3072 * sizeof(unsigned int));
    float*        biasAll = (float*)take(12 * 64 * sizeof(float));
    float*        cntNode = (float*)take((size_t)N * 4);
    float*        cntG    = (float*)take((size_t)G * 4);
    float*        gsum    = (float*)take((size_t)G * 64 * 4);
    _Float16*     gbuf    = (_Float16*)take((size_t)G * 64 * 2);
    _Float16*     gA      = (_Float16*)take((size_t)G * 64 * 2);
    _Float16*     gB      = (_Float16*)take((size_t)G * 64 * 2);
    _Float16*     gU      = (_Float16*)take((size_t)G * 64 * 2);
    float*        aggsum  = (float*)take((size_t)N * 64 * 4);
    _Float16*     n2in    = (_Float16*)take((size_t)N * 96 * 2);
    _Float16*     actNA   = (_Float16*)take((size_t)N * 64 * 2);
    _Float16*     actNB   = (_Float16*)take((size_t)N * 64 * 2);
    _Float16*     e0      = (_Float16*)take((size_t)E * 32 * 2);
    _Float16*     actA    = (_Float16*)take((size_t)E * 64 * 2);
    _Float16*     actB    = (_Float16*)take((size_t)E * 64 * 2);
    _Float16*     n1in    = (_Float16*)take((size_t)E * 96 * 2);

    // ---- zero accumulators (graph-capture-legal) ----
    hipMemsetAsync(stats,   0, 12 * 192 * sizeof(float), stream);
    hipMemsetAsync(cntNode, 0, (size_t)N * 4, stream);
    hipMemsetAsync(cntG,    0, (size_t)G * 4, stream);
    hipMemsetAsync(gsum,    0, (size_t)G * 64 * 4, stream);
    hipMemsetAsync(aggsum,  0, (size_t)N * 64 * 4, stream);

    // ---- launch helpers ----
    auto statsL = [&](int L, const _Float16* A, long rowsL, int stride, int nfeat) {
        dim3 grid(64, (unsigned)nfeat, 1);
        stats_cols<<<grid, 256, 0, stream>>>(A, rowsL, stride,
                                             stats + L * 192, stats + L * 192 + 96);
    };
    auto foldL = [&](int L, int K, int Kpad, long R) {
        int mb = 4 + (L / 3) * 12 + (L % 3) * 4;   // g, be, W, c per layer
        fold_bn<<<1, 256, 0, stream>>>(
            (const float*)d_in[mb + 2], (const float*)d_in[mb + 3],
            (const float*)d_in[mb + 0], (const float*)d_in[mb + 1],
            stats + L * 192, stats + L * 192 + 96, 1.0f / (float)R,
            K, Kpad, wpAll + L * 3072, biasAll + L * 64);
    };
    auto gemmL = [&](int L, const _Float16* A, long rowsL, int Kpad,
                     _Float16* outp, int ostr, int ooff, int relu,
                     float* accum, const int* sidx) {
        long tiles = (rowsL + 15) / 16;
        unsigned blocks = (unsigned)((tiles + 7) / 8);
        gemm64<<<blocks, 256, 0, stream>>>(A, rowsL, Kpad, wpAll + L * 3072,
                                           biasAll + L * 64, outp, ostr, ooff, relu,
                                           accum, sidx);
    };
    auto blocks1d = [](long n) { return (unsigned)((n + 255) / 256); };

    // ---- stage 0: gathers + segment counts ----
    gather_edge_in<<<blocks1d(E), 256, 0, stream>>>(x, ea, row, col, E, e0, n1in);
    gather_node_in<<<blocks1d(N), 256, 0, stream>>>(x, N, n2in);
    count_seg<<<blocks1d(E), 256, 0, stream>>>(col, E, cntNode);
    count_seg<<<blocks1d(N), 256, 0, stream>>>(batch, N, cntG);

    // ---- edge MLP (layers 0..2), rows = E ----
    statsL(0, e0, E, 32, 30);   foldL(0, 30, 32, E);
    gemmL(0, e0, E, 32, actA, 64, 0, 1, nullptr, nullptr);
    statsL(1, actA, E, 64, 64); foldL(1, 64, 64, E);
    gemmL(1, actA, E, 64, actB, 64, 0, 1, nullptr, nullptr);
    statsL(2, actB, E, 64, 64); foldL(2, 64, 64, E);
    gemmL(2, actB, E, 64, n1in, 96, 9, 0, nullptr, nullptr);   // e -> n1 concat cols 9..72

    // ---- node MLP 1 (layers 3..5), rows = E; last layer scatter-sums by col ----
    statsL(3, n1in, E, 96, 73); foldL(3, 73, 96, E);
    gemmL(3, n1in, E, 96, actA, 64, 0, 1, nullptr, nullptr);
    statsL(4, actA, E, 64, 64); foldL(4, 64, 64, E);
    gemmL(4, actA, E, 64, actB, 64, 0, 1, nullptr, nullptr);
    statsL(5, actB, E, 64, 64); foldL(5, 64, 64, E);
    gemmL(5, actB, E, 64, nullptr, 0, 0, 0, aggsum, col);      // h scatter-add -> aggsum

    // agg = seg_mean -> n2 concat cols 9..72
    div_seg<<<blocks1d(N * 64), 256, 0, stream>>>(aggsum, cntNode, N, n2in, 96, 9);

    // ---- node MLP 2 (layers 6..8), rows = N; last layer scatter-sums by batch ----
    statsL(6, n2in, N, 96, 73); foldL(6, 73, 96, N);
    gemmL(6, n2in, N, 96, actNA, 64, 0, 1, nullptr, nullptr);
    statsL(7, actNA, N, 64, 64); foldL(7, 64, 64, N);
    gemmL(7, actNA, N, 64, actNB, 64, 0, 1, nullptr, nullptr);
    statsL(8, actNB, N, 64, 64); foldL(8, 64, 64, N);
    gemmL(8, actNB, N, 64, nullptr, 0, 0, 0, gsum, batch);     // xn scatter-add -> gsum

    // g = seg_mean -> gbuf [G,64]
    div_seg<<<blocks1d((long)G * 64), 256, 0, stream>>>(gsum, cntG, G, gbuf, 64, 0);

    // ---- global MLP (layers 9..11), rows = G ----
    statsL(9,  gbuf, G, 64, 64); foldL(9, 64, 64, G);
    gemmL(9,  gbuf, G, 64, gA, 64, 0, 1, nullptr, nullptr);
    statsL(10, gA,  G, 64, 64); foldL(10, 64, 64, G);
    gemmL(10, gA,  G, 64, gB, 64, 0, 1, nullptr, nullptr);
    statsL(11, gB,  G, 64, 64); foldL(11, 64, 64, G);
    gemmL(11, gB,  G, 64, gU, 64, 0, 0, nullptr, nullptr);

    // ---- prediction + softmax ----
    pred_softmax<<<1, 256, 0, stream>>>(gU, (const float*)d_in[52], (const float*)d_in[53],
                                        (float*)d_out, G);
    (void)n_in; (void)out_size; (void)ws_size;
}
